// CausalMultiheadAttention_62062277427345
// MI455X (gfx1250) — matrix operand order repaired and checked by reading, not compiled, not run on hardware
//
#include <hip/hip_runtime.h>

// ---------------------------------------------------------------------------
// CausalMultiheadAttention decode step, MI455X (gfx1250, wave32, WMMA)
// K/V projections: bf16 WMMA, double-buffered LDS, async B staging.
// ---------------------------------------------------------------------------

typedef __attribute__((ext_vector_type(16))) __bf16        v16bf;
typedef __attribute__((ext_vector_type(2)))  __bf16        v2bf;
typedef __attribute__((ext_vector_type(8)))  float         v8f;
typedef __attribute__((ext_vector_type(8)))  unsigned int  v8u;
typedef __attribute__((ext_vector_type(4)))  int           v4i;

constexpr int kNHID  = 1024;
constexpr int kHeads = 16;
constexpr int kHD    = 64;     // head dim
constexpr int kB     = 32;     // batch
constexpr int kL     = 4096;   // cache length
constexpr float kScale = 0.125f; // 1/sqrt(64)

// ---- gfx1250 async load-to-LDS availability -------------------------------
#if defined(__has_builtin)
#  if __has_builtin(__builtin_amdgcn_global_load_async_to_lds_b128)
#    define HAVE_ASYNC_LDS 1
#  endif
#endif
#ifndef HAVE_ASYNC_LDS
#  define HAVE_ASYNC_LDS 0
#endif

#if HAVE_ASYNC_LDS
#  if __has_builtin(__builtin_amdgcn_s_wait_asynccnt)
#    define ASYNC_WAIT0() __builtin_amdgcn_s_wait_asynccnt(0)
#  else
#    define ASYNC_WAIT0() asm volatile("s_wait_asynccnt 0x0" ::: "memory")
#  endif
#else
#  define ASYNC_WAIT0()
#endif

// f32 -> bf16 using native converts; paired version keeps the packed result
// of v_cvt_pk_bf16_f32 intact (no re-extract / re-pack VALU ops).
__device__ __forceinline__ unsigned short f2bf(float f) {
    __bf16 h = (__bf16)f;
    return __builtin_bit_cast(unsigned short, h);
}
__device__ __forceinline__ unsigned int f2bf2(float a, float b) {
    v2bf p; p[0] = (__bf16)a; p[1] = (__bf16)b;
    return __builtin_bit_cast(unsigned int, p);
}
__device__ __forceinline__ float bf2f_lo(unsigned int packed) {
    return __uint_as_float(packed << 16);
}
__device__ __forceinline__ float bf2f_hi(unsigned int packed) {
    return __uint_as_float(packed & 0xffff0000u);
}

// ---------------------------------------------------------------------------
// One-time weight conversion f32 -> bf16 (weights are L2-hot; done once
// instead of once per output tile block).
// ---------------------------------------------------------------------------
__global__ __launch_bounds__(256)
void conv_bf16(const float* __restrict__ src, unsigned short* __restrict__ dst,
               int n4) {
    const int gid = blockIdx.x * 256 + threadIdx.x;
    if (gid >= n4) return;
    float4 a = ((const float4*)src)[gid];
    uint2 p; p.x = f2bf2(a.x, a.y); p.y = f2bf2(a.z, a.w);
    *(uint2*)(dst + (size_t)gid * 4) = p;
}

// ---------------------------------------------------------------------------
// Tiled GEMM:  Out[M,N](bf16) = X[M,K](f32) @ Wbf[N,K]^T(bf16) + bias
// 128x128 tile / 256 threads (8 waves), K-step 32, double-buffered LDS.
// A tile: f32 global -> regs -> bf16 LDS (converted in flight, overlapped
// with WMMA). B tile: bf16 global -> LDS via async load-to-LDS (ASYNCcnt).
// Causally dead blocks (cache position > position_i) exit immediately.
// ---------------------------------------------------------------------------
#define TM 128
#define TN 128
#define TK 32
#define LDT 40   // padded LDS row stride (bf16 elems): 80B, conflict-free

__global__ __launch_bounds__(256)
void proj_gemm_wmma(const float* __restrict__ X,            // [M, K] f32
                    const unsigned short* __restrict__ Wbf, // [N, K] bf16
                    const float* __restrict__ bias,         // [N]
                    unsigned short* __restrict__ Out,       // [M, N] bf16
                    int Kdim,
                    const int* __restrict__ pos_p,
                    int maskL)                              // 0 = no causal skip
{
    const int m0 = blockIdx.x * TM;
    const int n0 = blockIdx.y * TN;
    if (maskL > 0) {
        const int pos = *pos_p;           // uniform scalar load
        if ((m0 % maskL) > pos) return;   // fully masked rows: dead work
    }

    __shared__ unsigned short sA[2][TM * LDT];
    __shared__ unsigned short sB[2][TN * LDT];

    const int tid  = threadIdx.x;
    const int lane = tid & 31;
    const int w    = tid >> 5;
    const int wm   = (w & 3) * 32;        // wave's M offset inside tile
    const int wn   = (w >> 2) * 64;       // wave's N offset inside tile

    const float* Xblk = X + (size_t)m0 * Kdim;

    // ---- staging helpers ----
    float4 aregs[4];
    auto loadA = [&](int kk) {
        #pragma unroll
        for (int r = 0; r < 4; ++r) {
            const int s   = tid + 256 * r;     // 1024 float4 slots
            const int row = s >> 3;
            const int c4  = (s & 7) << 2;
            aregs[r] = *(const float4*)(Xblk + (size_t)row * Kdim + kk + c4);
        }
    };
    auto storeA = [&](int buf) {
        #pragma unroll
        for (int r = 0; r < 4; ++r) {
            const int s   = tid + 256 * r;
            const int row = s >> 3;
            const int c4  = (s & 7) << 2;
            // 4 bf16 packed as 2 dwords -> single ds_store_b64
            uint2 p;
            p.x = f2bf2(aregs[r].x, aregs[r].y);
            p.y = f2bf2(aregs[r].z, aregs[r].w);
            *(uint2*)&sA[buf][row * LDT + c4] = p;
        }
    };
#if HAVE_ASYNC_LDS
    auto stageB = [&](int kk, int buf) {
        #pragma unroll
        for (int r = 0; r < 2; ++r) {
            const int s   = tid + 256 * r;     // 512 x 16B chunks
            const int row = s >> 2;
            const int c16 = (s & 3) << 3;      // bf16 elem offset
            const unsigned short* g = Wbf + (size_t)(n0 + row) * Kdim + kk + c16;
            unsigned short* l = &sB[buf][row * LDT + c16];
            __builtin_amdgcn_global_load_async_to_lds_b128((v4i*)g, (v4i*)l, 0, 0);
        }
    };
#else
    uint4 bregs[2];
    auto loadB = [&](int kk) {
        #pragma unroll
        for (int r = 0; r < 2; ++r) {
            const int s   = tid + 256 * r;
            const int row = s >> 2;
            const int c16 = (s & 3) << 3;
            bregs[r] = *(const uint4*)(Wbf + (size_t)(n0 + row) * Kdim + kk + c16);
        }
    };
    auto storeB = [&](int buf) {
        #pragma unroll
        for (int r = 0; r < 2; ++r) {
            const int s   = tid + 256 * r;
            const int row = s >> 2;
            const int c16 = (s & 3) << 3;
            *(uint4*)&sB[buf][row * LDT + c16] = bregs[r];
        }
    };
#endif

    v8f acc[2][4] = {};

    // ---- prologue: stage tile 0 into buffer 0 ----
    loadA(0);
#if HAVE_ASYNC_LDS
    stageB(0, 0);
#else
    loadB(0);
    storeB(0);
#endif
    storeA(0);
    ASYNC_WAIT0();
    __syncthreads();

    const int nk = Kdim / TK;
    for (int t = 0; t < nk; ++t) {
        const int cur = t & 1;
        const int nxt = cur ^ 1;
        const bool hasNext = (t + 1) < nk;

        // prefetch next tile (overlaps with WMMA below)
        if (hasNext) {
            loadA((t + 1) * TK);
#if HAVE_ASYNC_LDS
            stageB((t + 1) * TK, nxt);
#else
            loadB((t + 1) * TK);
#endif
        }

        // ---- fragments from current buffer ----
        // A 16x32 bf16: lanes 0-15 row m, K 0..7 & 16..23; lanes 16-31 K 8..15 & 24..31
        v16bf afrag[2];
        {
            const int m  = lane & 15;
            const int kb = (lane >> 4) << 3;
            #pragma unroll
            for (int i = 0; i < 2; ++i) {
                const unsigned short* base = &sA[cur][(wm + i * 16 + m) * LDT];
                uint4 lo = *(const uint4*)(base + kb);
                uint4 hi = *(const uint4*)(base + kb + 16);
                v8u t8;
                t8[0] = lo.x; t8[1] = lo.y; t8[2] = lo.z; t8[3] = lo.w;
                t8[4] = hi.x; t8[5] = hi.y; t8[6] = hi.z; t8[7] = hi.w;
                afrag[i] = __builtin_bit_cast(v16bf, t8);
            }
        }
        // B 32x16 bf16: column n = lane&15, lanes<16: K 0..7 & 16..23; lanes>=16: K 8..15 & 24..31
        v16bf bfrag[4];
        {
            const int n  = lane & 15;
            const int kb = (lane < 16) ? 0 : 8;
            #pragma unroll
            for (int j = 0; j < 4; ++j) {
                const unsigned short* base = &sB[cur][(wn + j * 16 + n) * LDT + kb];
                uint4 lo = *(const uint4*)(base);
                uint4 hi = *(const uint4*)(base + 16);
                v8u t8;
                t8[0] = lo.x; t8[1] = lo.y; t8[2] = lo.z; t8[3] = lo.w;
                t8[4] = hi.x; t8[5] = hi.y; t8[6] = hi.z; t8[7] = hi.w;
                bfrag[j] = __builtin_bit_cast(v16bf, t8);
            }
        }

        #pragma unroll
        for (int i = 0; i < 2; ++i)
            #pragma unroll
            for (int j = 0; j < 4; ++j)
                acc[i][j] = __builtin_amdgcn_wmma_f32_16x16x32_bf16(
                    false, afrag[i], false, bfrag[j],
                    (short)0, acc[i][j], false, false);

        // write prefetched A (and fallback B) into the other buffer
        if (hasNext) {
            storeA(nxt);
#if !HAVE_ASYNC_LDS
            storeB(nxt);
#endif
        }
        ASYNC_WAIT0();
        __syncthreads();
    }

    // ---- epilogue: +bias, store bf16. C layout: vgpr r = row r (lanes<16)
    //      or row 8+r (lanes>=16); col = lane&15 ----
    #pragma unroll
    for (int j = 0; j < 4; ++j) {
        const int col = n0 + wn + j * 16 + (lane & 15);
        const float bv = bias[col];
        #pragma unroll
        for (int i = 0; i < 2; ++i) {
            const int rb = m0 + wm + i * 16 + ((lane >> 4) << 3);
            #pragma unroll
            for (int r = 0; r < 8; ++r)
                Out[(size_t)(rb + r) * kNHID + col] = f2bf(acc[i][j][r] + bv);
        }
    }
}

// ---------------------------------------------------------------------------
// Small GEMV projection: Out[b,n] = X[b,:]·W[n,:] + bias[n]   (f32)
// Q projection (32x1024) and output projection. Weight rows L2-hot.
// ---------------------------------------------------------------------------
__global__ __launch_bounds__(256)
void proj_gemv(const float* __restrict__ X, const float* __restrict__ W,
               const float* __restrict__ bias, float* __restrict__ Out,
               int Brows, int N, int Kdim)
{
    const int gid = blockIdx.x * 256 + threadIdx.x;
    if (gid >= Brows * N) return;
    const int b = gid / N;
    const int n = gid - b * N;
    const float4* x4 = (const float4*)(X + (size_t)b * Kdim);
    const float4* w4 = (const float4*)(W + (size_t)n * Kdim);
    float s = 0.f;
    for (int i = 0; i < Kdim / 4; ++i) {
        float4 a = x4[i], c = w4[i];
        s = fmaf(a.x, c.x, s); s = fmaf(a.y, c.y, s);
        s = fmaf(a.z, c.z, s); s = fmaf(a.w, c.w, s);
    }
    Out[gid] = s + bias[n];
}

// ---------------------------------------------------------------------------
// Attention decode: one block per (batch, head). Scores for all unmasked
// positions live in LDS (16 KB); classic max/sum block reductions.
// ---------------------------------------------------------------------------
__global__ __launch_bounds__(256)
void attn_decode(const unsigned short* __restrict__ Kc,  // [B, L, NHID] bf16
                 const unsigned short* __restrict__ Vc,  // [B, L, NHID] bf16
                 const float* __restrict__ Q,            // [B, NHID] f32
                 const int* __restrict__ pos_p,
                 float* __restrict__ Out)                 // [B, NHID] f32
{
    const int bh = blockIdx.x;
    const int b  = bh >> 4;
    const int h  = bh & 15;
    int pos = *pos_p;
    if (pos >= kL) pos = kL - 1;
    if (pos < 0)  pos = 0;

    __shared__ float sq[kHD];
    __shared__ float sc[kL];
    __shared__ float red[256];

    const int tid = threadIdx.x;
    if (tid < kHD) sq[tid] = Q[(size_t)b * kNHID + h * kHD + tid];
    __syncthreads();

    // scores: s[l] = k[b,l,h,:] · q[b,h,:]
    float lmax = -3.0e38f;
    for (int l = tid; l <= pos; l += 256) {
        const uint4* kr = (const uint4*)(Kc + ((size_t)b * kL + l) * kNHID + h * kHD);
        float s = 0.f;
        #pragma unroll
        for (int c = 0; c < 8; ++c) {        // 8 * uint4 = 64 bf16
            uint4 u = kr[c];
            unsigned int uu[4] = {u.x, u.y, u.z, u.w};
            #pragma unroll
            for (int q4 = 0; q4 < 4; ++q4) {
                s = fmaf(bf2f_lo(uu[q4]), sq[c * 8 + q4 * 2 + 0], s);
                s = fmaf(bf2f_hi(uu[q4]), sq[c * 8 + q4 * 2 + 1], s);
            }
        }
        sc[l] = s;
        lmax = fmaxf(lmax, s);
    }
    red[tid] = lmax; __syncthreads();
    for (int s2 = 128; s2 > 0; s2 >>= 1) {
        if (tid < s2) red[tid] = fmaxf(red[tid], red[tid + s2]);
        __syncthreads();
    }
    const float gmax = red[0] * kScale;   // (s + 0)*scale for unmasked positions
    __syncthreads();

    float lsum = 0.f;
    for (int l = tid; l <= pos; l += 256) {
        float p = __expf(sc[l] * kScale - gmax);
        sc[l] = p;
        lsum += p;
    }
    red[tid] = lsum; __syncthreads();
    for (int s2 = 128; s2 > 0; s2 >>= 1) {
        if (tid < s2) red[tid] += red[tid + s2];
        __syncthreads();
    }
    const float inv = 1.0f / red[0];
    __syncthreads();

    // out[d] = sum_l p[l] * v[b,l,h,d] : 64 dims x 4 l-strided partials
    const int d = tid & 63;
    const int g = tid >> 6;
    float acc = 0.f;
    for (int l = g; l <= pos; l += 4) {
        unsigned int hv = Vc[((size_t)b * kL + l) * kNHID + h * kHD + d];
        acc = fmaf(sc[l], __uint_as_float(hv << 16), acc);
    }
    red[tid] = acc; __syncthreads();
    if (tid < 64) {
        Out[(size_t)b * kNHID + h * kHD + tid] =
            (red[tid] + red[64 + tid] + red[128 + tid] + red[192 + tid]) * inv;
    }
}

// ---------------------------------------------------------------------------
// Launch
// ---------------------------------------------------------------------------
extern "C" void kernel_launch(void* const* d_in, const int* in_sizes, int n_in,
                              void* d_out, int out_size, void* d_ws, size_t ws_size,
                              hipStream_t stream) {
    const float* query = (const float*)d_in[0];
    const float* key_c = (const float*)d_in[1];
    const float* val_c = (const float*)d_in[2];
    const int*   pos_p = (const int*)  d_in[3];
    const float* Wq = (const float*)d_in[4];  const float* bq = (const float*)d_in[5];
    const float* Wk = (const float*)d_in[6];  const float* bk = (const float*)d_in[7];
    const float* Wv = (const float*)d_in[8];  const float* bv = (const float*)d_in[9];
    const float* Wo = (const float*)d_in[10]; const float* bo = (const float*)d_in[11];

    // workspace layout
    float* q_ws    = (float*)d_ws;                       // 32*1024 f32
    float* attn_ws = q_ws + kB * kNHID;                  // 32*1024 f32
    unsigned short* wk_bf = (unsigned short*)(attn_ws + kB * kNHID); // 2 MB
    unsigned short* wv_bf = wk_bf + (size_t)kNHID * kNHID;           // 2 MB
    unsigned short* k_ws  = wv_bf + (size_t)kNHID * kNHID;           // 256 MB
    unsigned short* v_ws  = k_ws + (size_t)kB * kL * kNHID;          // 256 MB

    const int w4 = kNHID * kNHID / 4;

    // 0) one-time weight conversion to bf16 (L2-hot, ~4 MB total)
    conv_bf16<<<dim3((w4 + 255) / 256), dim3(256), 0, stream>>>(Wk, wk_bf, w4);
    conv_bf16<<<dim3((w4 + 255) / 256), dim3(256), 0, stream>>>(Wv, wv_bf, w4);

    // 1) Q projection (tiny)
    proj_gemv<<<dim3((kB * kNHID + 255) / 256), dim3(256), 0, stream>>>(
        query, Wq, bq, q_ws, kB, kNHID, kNHID);

    // 2) K and V projections (the ~550 GFLOP bulk) — bf16 WMMA, causal skip
    dim3 ggrid((kB * kL) / TM, kNHID / TN);
    proj_gemm_wmma<<<ggrid, dim3(256), 0, stream>>>(key_c, wk_bf, bk, k_ws, kNHID, pos_p, kL);
    proj_gemm_wmma<<<ggrid, dim3(256), 0, stream>>>(val_c, wv_bf, bv, v_ws, kNHID, pos_p, kL);

    // 3) attention: one block per (batch, head)
    attn_decode<<<dim3(kB * kHeads), dim3(256), 0, stream>>>(k_ws, v_ws, q_ws, pos_p, attn_ws);

    // 4) output projection
    proj_gemv<<<dim3((kB * kNHID + 255) / 256), dim3(256), 0, stream>>>(
        attn_ws, Wo, bo, (float*)d_out, kB, kNHID, kNHID);
}
